// ScaledDotProductAttention_11287174054066
// MI455X (gfx1250) — matrix operand order repaired and checked by reading, not compiled, not run on hardware
//
#include <hip/hip_runtime.h>

typedef __attribute__((ext_vector_type(16))) _Float16     v16h;
typedef __attribute__((ext_vector_type(8)))  float        v8f;
typedef __attribute__((ext_vector_type(4)))  unsigned int v4u;
typedef __attribute__((ext_vector_type(8)))  int          v8i_;
typedef __attribute__((ext_vector_type(4)))  int          v4i_;

#define B_       64
#define LQ_      1024
#define LK_      1024
#define D_       64
#define INV_TEMP 0.125f           // 1/sqrt(D) = 1/8
#define QTILE    16
#define S_STRIDE 1028             // 1024+4: row-to-row bank offset 4 -> 16 rows conflict-free
#define VSTRIDE  66               // 64+2: row-to-row bank offset 2 -> 32 rows conflict-free
#define NWAVES   4
#define NTHREADS 128

// ---- Tensor Data Mover: DMA one 64x64 f32 tile (row-major, stride 64) into LDS ----
// D# per CDNA5 ISA. LDS gets 2-dword padding every 64
// dwords (pad_interval code 5, pad_amount code 1) -> LDS row stride = 66 floats.
__device__ __forceinline__ void tdm_load_tile64(unsigned lds_byte_addr, const float* gptr) {
    unsigned long long ga = (unsigned long long)(uintptr_t)gptr;
    v4u g0;
    g0[0] = 1u;                                             // count=1, user desc
    g0[1] = lds_byte_addr;                                  // lds_addr
    g0[2] = (unsigned)(ga & 0xFFFFFFFFu);                   // global_addr[31:0]
    g0[3] = (unsigned)((ga >> 32) & 0x1FFFFFFu) | (2u << 30); // global_addr[56:32], type=2
    v8i_ g1;
    g1[0] = 0x03520000;          // wg_mask=0 | data_size=2(4B)<<16 | pad_en<<20 | intv=5<<22 | amt=1<<25
    g1[1] = (int)(64u << 16);    // tensor_dim0 = 64  (bits 79:48)
    g1[2] = (int)(64u << 16);    // tensor_dim0_hi=0, tensor_dim1 = 64 (bits 111:80)
    g1[3] = (int)(64u << 16);    // tensor_dim1_hi=0, tile_dim0 = 64 (bits 127:112)
    g1[4] = 64;                  // tile_dim1 = 64, tile_dim2 = 0
    g1[5] = 64;                  // tensor_dim0_stride = 64 (bits 207:160, low dword)
    g1[6] = 0;                   // stride_hi = 0, tensor_dim1_stride = 0 (2-D tile)
    g1[7] = 0;
    v4i_ z4 = {0, 0, 0, 0};
    v8i_ z8 = {0, 0, 0, 0, 0, 0, 0, 0};
    __builtin_amdgcn_tensor_load_to_lds(g0, g1, z4, z4, z8, 0);   // 6-arg form
}

__global__ __launch_bounds__(NTHREADS)
void sdpa_fused_kernel(const float* __restrict__ q,
                       const float* __restrict__ k,
                       const float* __restrict__ v,
                       const unsigned char* __restrict__ qmask,   // [B,LQ]
                       const unsigned char* __restrict__ kmask,   // [B,LK]
                       const unsigned char* __restrict__ gmask,   // [B,LQ,LK]
                       float* __restrict__ out_av,                // [B,LQ,D]
                       float* __restrict__ out_attn)              // [B,LQ,LK]
{
    __shared__ float S[QTILE * S_STRIDE];
    __shared__ float rowMax[QTILE];
    __shared__ float rowScale[QTILE];
    __shared__ alignas(256) float Stage[2][64 * VSTRIDE];   // K (pass1) / V (pass2) staging

    const int tid  = threadIdx.x;
    const int lane = tid & 31;
    const int wave = tid >> 5;
    const int b    = blockIdx.x >> 6;
    const int q0   = (blockIdx.x & 63) * QTILE;

    const int m  = lane & 15;
    const int hi = lane >> 4;

    // ---- A fragments: Q tile (16x64), 16-bit A 16x32 layout (ISA 7.12.2) ----
    v16h a0, a1;
    {
        const float* qrow = q + (size_t)(b * LQ_ + q0 + m) * D_;
#pragma unroll
        for (int j = 0; j < 16; ++j) {
            const int vv = j >> 1, pos = j & 1;
            const int kk = ((vv & 4) ? 16 : 0) + 2 * (vv & 3) + hi * 8 + pos;  // 0..31
            a0[j] = (_Float16)qrow[kk];
            a1[j] = (_Float16)qrow[32 + kk];
        }
    }

    const int klane = (lane & 15) + (hi << 4);   // unused (equals lane); B column owned by this lane is lane & 15
    const unsigned stage_base = (unsigned)(uintptr_t)(void*)&Stage[0][0];
    const unsigned stage_sz   = 64 * VSTRIDE * 4;

    // ================= pass 1: S = mask(Q K^T * invT) -> LDS =================
    // 16 iterations, each stages 64 K-rows via TDM; wave w computes cols w*16..w*16+15.
    if (wave == 0)
        tdm_load_tile64(stage_base, k + (size_t)(b * LK_) * D_);

    for (int it = 0; it < LK_ / 64; ++it) {
        const int cur = it & 1;
        if (wave == 0) {
            if (it + 1 < LK_ / 64) {
                tdm_load_tile64(stage_base + (cur ^ 1) * stage_sz,
                                k + (size_t)(b * LK_ + (it + 1) * 64) * D_);
                __builtin_amdgcn_s_wait_tensorcnt(1);      // tile `it` complete (in-order)
            } else {
                __builtin_amdgcn_s_wait_tensorcnt(0);
            }
        }
        __syncthreads();                                   // Stage[cur] holds K rows

        const float* Kst = &Stage[cur][0];
        v16h b0, b1;
#pragma unroll
        for (int j = 0; j < 16; ++j) {
            const int vv = j >> 1, pos = j & 1;
            const int kk = ((vv & 4) ? 16 : 0) + 2 * (vv & 3) + hi * 8 + pos;  // 0..31
            const float* krow = Kst + (wave * 16 + m) * VSTRIDE;
            b0[j] = (_Float16)krow[kk];                    // B[k][n] = K[n][k]
            b1[j] = (_Float16)krow[32 + kk];
        }
        v8f c = {};
        c = __builtin_amdgcn_wmma_f32_16x16x32_f16(false, a0, false, b0, (short)0, c, false, false);
        c = __builtin_amdgcn_wmma_f32_16x16x32_f16(false, a1, false, b1, (short)0, c, false, false);

        const int  ncol = it * 64 + wave * 16 + (lane & 15);
        const bool kmq  = kmask[b * LK_ + ncol] != 0;
#pragma unroll
        for (int r = 0; r < 8; ++r) {
            const int  row = r + hi * 8;
            const bool gmq = gmask[(size_t)(b * LQ_ + q0 + row) * LK_ + ncol] != 0;
            S[row * S_STRIDE + ncol] = (kmq && gmq) ? c[r] * INV_TEMP : -__builtin_inff();
        }
        __syncthreads();                                   // done with Stage[cur]
    }

    // kick off V(0) DMA; it overlaps the softmax phase below
    if (wave == 0)
        tdm_load_tile64(stage_base, v + (size_t)(b * LK_) * D_);

    // ================= softmax stats: wave owns 4 rows =================
    for (int rr = 0; rr < 4; ++rr) {
        const int row = wave * 4 + rr;
        const float* srow = &S[row * S_STRIDE];

        float mx = -__builtin_inff();
        for (int c0 = lane; c0 < LK_; c0 += 32) mx = fmaxf(mx, srow[c0]);
#pragma unroll
        for (int off = 16; off > 0; off >>= 1) mx = fmaxf(mx, __shfl_xor(mx, off, 32));

        float sum = 0.f;
        for (int c0 = lane; c0 < LK_; c0 += 32) {
            const float s = srow[c0];
            sum += (s == -__builtin_inff()) ? 0.f : __expf(s - mx);
        }
#pragma unroll
        for (int off = 16; off > 0; off >>= 1) sum += __shfl_xor(sum, off, 32);

        if (lane == 0) {
            const bool qm = qmask[b * LQ_ + q0 + row] != 0;
            rowMax[row]   = mx;
            rowScale[row] = (qm && sum > 0.f) ? (1.0f / sum) : 0.0f;   // query mask folded in
        }
    }
    __syncthreads();

    // ---- normalize in LDS + single streaming write of attn ----
    for (int idx = tid; idx < QTILE * LK_; idx += NTHREADS) {
        const int   row = idx >> 10;
        const int   col = idx & (LK_ - 1);
        const float s   = S[row * S_STRIDE + col];
        const float p   = (s == -__builtin_inff()) ? 0.f
                          : __expf(s - rowMax[row]) * rowScale[row];
        S[row * S_STRIDE + col] = p;
        out_attn[(size_t)(b * LQ_ + q0 + row) * LK_ + col] = p;
    }
    __syncthreads();

    // ================= pass 2: O = P * V ; wave owns D-slice n0..n0+15 =================
    const int n0 = wave * 16;
    v8f o = {};
    for (int it = 0; it < LK_ / 64; ++it) {
        const int cur = it & 1;
        if (wave == 0) {
            if (it + 1 < LK_ / 64) {
                tdm_load_tile64(stage_base + (cur ^ 1) * stage_sz,
                                v + (size_t)(b * LK_ + (it + 1) * 64) * D_);
                __builtin_amdgcn_s_wait_tensorcnt(1);      // tile `it` complete (in-order)
            } else {
                __builtin_amdgcn_s_wait_tensorcnt(0);
            }
        }
        __syncthreads();                                   // Stage[cur] holds 64 V rows

        const float* Vst = &Stage[cur][0];
        const int kk0 = it * 64;
        v16h ap0, ap1, bv0, bv1;
#pragma unroll
        for (int j = 0; j < 16; ++j) {
            const int vv = j >> 1, pos = j & 1;
            const int kl = ((vv & 4) ? 16 : 0) + 2 * (vv & 3) + hi * 8 + pos;  // 0..31
            ap0[j] = (_Float16)S[m * S_STRIDE + kk0 + kl];
            ap1[j] = (_Float16)S[m * S_STRIDE + kk0 + 32 + kl];
            bv0[j] = (_Float16)Vst[kl * VSTRIDE + n0 + m];             // V rows kk0..kk0+31
            bv1[j] = (_Float16)Vst[(32 + kl) * VSTRIDE + n0 + m];      // V rows kk0+32..kk0+63
        }
        o = __builtin_amdgcn_wmma_f32_16x16x32_f16(false, ap0, false, bv0, (short)0, o, false, false);
        o = __builtin_amdgcn_wmma_f32_16x16x32_f16(false, ap1, false, bv1, (short)0, o, false, false);
        __syncthreads();                                   // done with Stage[cur]
    }

#pragma unroll
    for (int r = 0; r < 8; ++r) {
        const int row = r + hi * 8;
        out_av[(size_t)(b * LQ_ + q0 + row) * D_ + n0 + (lane & 15)] = o[r];
    }
}

extern "C" void kernel_launch(void* const* d_in, const int* in_sizes, int n_in,
                              void* d_out, int out_size, void* d_ws, size_t ws_size,
                              hipStream_t stream) {
    const float* q  = (const float*)d_in[0];
    const float* k  = (const float*)d_in[1];
    const float* v  = (const float*)d_in[2];
    const unsigned char* qm = (const unsigned char*)d_in[3];
    const unsigned char* km = (const unsigned char*)d_in[4];
    const unsigned char* gm = (const unsigned char*)d_in[5];

    float* out_av   = (float*)d_out;                       // [B,LQ,D] first (tuple order)
    float* out_attn = out_av + (size_t)B_ * LQ_ * D_;      // then [B,LQ,LK]

    dim3 grid(B_ * (LQ_ / QTILE));                         // 4096 workgroups
    sdpa_fused_kernel<<<grid, NTHREADS, 0, stream>>>(q, k, v, qm, km, gm, out_av, out_attn);
}
